// Attention_58909771432790
// MI455X (gfx1250) — matrix operand order repaired and checked by reading, not compiled, not run on hardware
//
#include <hip/hip_runtime.h>
#include <hip/hip_bf16.h>
#include <stdint.h>

#define DIM 4096
#define SEQ 1024
#define NH 32
#define HD 128

typedef _Float16 v16h __attribute__((ext_vector_type(16)));
typedef _Float16 v8h  __attribute__((ext_vector_type(8)));
typedef float    v8f  __attribute__((ext_vector_type(8)));
typedef int      v4i  __attribute__((ext_vector_type(4)));

#if defined(__has_builtin)
#if __has_builtin(__builtin_amdgcn_global_load_async_to_lds_b128)
#define HAVE_ASYNC_LDS 1
#endif
#endif

#ifdef HAVE_ASYNC_LDS
// Builtin expects int4 pointers: global (addrspace 1) src, LDS (addrspace 3) dst.
typedef v4i __attribute__((address_space(1))) as1_v4i;
typedef v4i __attribute__((address_space(3))) as3_v4i;
#if __has_builtin(__builtin_amdgcn_s_wait_asynccnt)
#define WAIT_ASYNC0() __builtin_amdgcn_s_wait_asynccnt(0)
#else
#define WAIT_ASYNC0() asm volatile("s_wait_asynccnt 0" ::: "memory")
#endif
#endif

// Build a 16-half fragment from two 16B-aligned 8-half chunks.
__device__ __forceinline__ v16h ld8x2(const _Float16* p0, const _Float16* p1) {
  v8h a = *(const v8h*)p0;
  v8h b = *(const v8h*)p1;
  v16h r;
#pragma unroll
  for (int i = 0; i < 8; ++i) { r[i] = a[i]; r[i + 8] = b[i]; }
  return r;
}

// Dequantize 16 Q4_0 nibbles (one per byte position) from a 16-byte chunk.
// sh = 24 -> high nibble (even group), sh = 28 -> low nibble (odd group).
__device__ __forceinline__ v16h dq16(const uint4 w, const float s, const int sh) {
  const uint32_t wa[4] = {w.x, w.y, w.z, w.w};
  v16h r;
#pragma unroll
  for (int t = 0; t < 16; ++t) {
    const uint32_t b = wa[t >> 2] >> ((t & 3) * 8);
    const int q = ((int)(b << sh)) >> 28;
    r[t] = (_Float16)((float)q * s);
  }
  return r;
}

__global__ __launch_bounds__(256) void cvt_kernel(const float* __restrict__ x,
                                                  _Float16* __restrict__ xh, int n) {
  int i = blockIdx.x * 256 + threadIdx.x;
  if (i < n) xh[i] = (_Float16)x[i];
}

// out[m, j] = sum_k X[m,k] * deq(W)[j,k]
// Q4_0: flat f = j*DIM + k; group g = f>>6; bytes at (g>>1)*64 + (f&63);
// even g -> high nibble, odd g -> low nibble. For a lane's 16-k slice at
// r = kk + hl*16 the chunk at j*2048 + kc/2 + (r&63) serves kk and kk+64.
// Block: 256 thr (8 waves). Tile: 128 rows x 128 cols; wave owns 16 cols x 128 rows.
__global__ __launch_bounds__(256)
void gemm_q40_kernel(const _Float16* __restrict__ X,
                     const int8_t* __restrict__ Wp,
                     const float* __restrict__ Sc,
                     _Float16* __restrict__ Oh,
                     float* __restrict__ Of) {
  __shared__ _Float16 As[128][136];         // padded: stride 272B, 16B-aligned chunks
  const int lane = threadIdx.x & 31;
  const int wv   = threadIdx.x >> 5;
  const int hl   = lane >> 4;               // lane-half (wave32)
  const int l15  = lane & 15;
  const int rowb = blockIdx.y * 128;
  const int j    = blockIdx.x * 128 + wv * 16 + l15;   // output column == weight row
  const uint8_t* wbase = (const uint8_t*)Wp + ((size_t)j << 11) + hl * 16;
  const int jG = j << 6;                    // scale row base = j*64
  v8f acc[8] = {};

  for (int kc = 0; kc < DIM; kc += 128) {
    __syncthreads();
#ifdef HAVE_ASYNC_LDS
    for (int t = threadIdx.x; t < 2048; t += 256) {
      const int r  = t >> 4;
      const int c8 = (t & 15) * 8;
      __builtin_amdgcn_global_load_async_to_lds_b128(
          (as1_v4i*)&X[(size_t)(rowb + r) * DIM + kc + c8],
          (as3_v4i*)&As[r][c8], 0, 0);
    }
    WAIT_ASYNC0();
#else
    for (int t = threadIdx.x; t < 2048; t += 256) {
      const int r  = t >> 4;
      const int c8 = (t & 15) * 8;
      *(v8h*)&As[r][c8] = *(const v8h*)&X[(size_t)(rowb + r) * DIM + kc + c8];
    }
#endif
    __syncthreads();

    // ---- weights: two aligned 16B chunks cover all four 32-k steps of this chunk
    const uint8_t* bp = wbase + (kc >> 1);
    __builtin_prefetch(bp + 64, 0, 1);      // next kc chunk -> global_prefetch_b8
    const uint4 w0 = *(const uint4*)bp;         // kk=0 (hi nibble), kk=64 (lo nibble)
    const uint4 w1 = *(const uint4*)(bp + 32);  // kk=32 (hi),       kk=96 (lo)
    const float s0 = Sc[jG + (kc >> 6)];
    const float s1 = Sc[jG + (kc >> 6) + 1];
    v16h bf[4];
    bf[0] = dq16(w0, s0, 24);
    bf[1] = dq16(w1, s0, 24);
    bf[2] = dq16(w0, s1, 28);
    bf[3] = dq16(w1, s1, 28);

#pragma unroll
    for (int kq = 0; kq < 4; ++kq) {
      const int kk = kq * 32;
#pragma unroll
      for (int mt = 0; mt < 8; ++mt) {
        const int r = mt * 16 + l15;
        v16h afrag = ld8x2(&As[r][kk + hl * 16], &As[r][kk + hl * 16 + 8]);
        acc[mt] = __builtin_amdgcn_wmma_f32_16x16x32_f16(
            false, afrag, false, bf[kq], (short)0, acc[mt], false, false);
      }
    }
  }

#pragma unroll
  for (int mt = 0; mt < 8; ++mt)
#pragma unroll
    for (int vi = 0; vi < 8; ++vi) {
      const int m = rowb + mt * 16 + hl * 8 + vi;   // C layout: lanes16-31 -> M+8
      const float v = acc[mt][vi];
      if (Oh) Oh[(size_t)m * DIM + j] = (_Float16)v;
      if (Of) Of[(size_t)m * DIM + j] = v;
    }
}

__global__ __launch_bounds__(256)
void rope_kernel(_Float16* __restrict__ qh, _Float16* __restrict__ kh,
                 const float* __restrict__ cosp, const float* __restrict__ sinp) {
  const int i = blockIdx.x * 256 + threadIdx.x;  // SEQ*NH*64 threads
  const int s = i >> 11;
  const int r = i & 2047;
  const int h = r >> 6;
  const int d = r & 63;
  const float c  = cosp[s * 64 + d];
  const float sn = sinp[s * 64 + d];
  const size_t b = (size_t)s * DIM + h * HD;
  const float q1 = (float)qh[b + d], q2 = (float)qh[b + 64 + d];
  qh[b + d]      = (_Float16)(q1 * c - q2 * sn);
  qh[b + 64 + d] = (_Float16)(q2 * c + q1 * sn);
  const float k1 = (float)kh[b + d], k2 = (float)kh[b + 64 + d];
  kh[b + d]      = (_Float16)(k1 * c - k2 * sn);
  kh[b + 64 + d] = (_Float16)(k2 * c + k1 * sn);
}

// Flash attention: one head per block, 128 q rows (8 waves x 16), causal,
// online softmax, key blocks of 32. All matmuls via v_wmma_f32_16x16x32_f16.
__global__ __launch_bounds__(256)
void attn_kernel(const _Float16* __restrict__ Q, const _Float16* __restrict__ K,
                 const _Float16* __restrict__ V, _Float16* __restrict__ O) {
  __shared__ _Float16 Ks[32][136];     // [key][d], padded
  __shared__ _Float16 Vt[128][32];     // [d][key] (transposed at stage time)
  __shared__ _Float16 Ps[8][16][32];   // per-wave P staging (C-frag -> A-frag)
  const int h    = blockIdx.y;
  const int qb   = blockIdx.x * 128;
  const int lane = threadIdx.x & 31;
  const int wv   = threadIdx.x >> 5;
  const int hl   = lane >> 4;
  const int l15  = lane & 15;
  const int qw   = qb + wv * 16;

  // Q A-fragments for this wave's 16 rows over d=0..127 (kept in VGPRs).
  v16h aq[4];
#pragma unroll
  for (int kf = 0; kf < 4; ++kf) {
    const _Float16* p = Q + (size_t)(qw + l15) * DIM + h * HD + kf * 32;
    aq[kf] = ld8x2(p + hl * 16, p + hl * 16 + 8);
  }

  v8f ov[8] = {};
  float mrow[8], lrow[8];
#pragma unroll
  for (int vi = 0; vi < 8; ++vi) { mrow[vi] = -3.0e38f; lrow[vi] = 0.f; }
  const float sc = 0.08838834764831845f;  // 1/sqrt(128)
  const int kend = qb + 128;

  for (int kb = 0; kb < kend; kb += 32) {
    __syncthreads();
#ifdef HAVE_ASYNC_LDS
    for (int t = threadIdx.x; t < 512; t += 256) {
      const int key = t >> 4;
      const int c8  = (t & 15) * 8;
      __builtin_amdgcn_global_load_async_to_lds_b128(
          (as1_v4i*)&K[(size_t)(kb + key) * DIM + h * HD + c8],
          (as3_v4i*)&Ks[key][c8], 0, 0);
    }
    for (int t = threadIdx.x; t < 512; t += 256) {
      const int key = t >> 4;
      const int c8  = (t & 15) * 8;
      const v8h vv = *(const v8h*)&V[(size_t)(kb + key) * DIM + h * HD + c8];
#pragma unroll
      for (int i = 0; i < 8; ++i) Vt[c8 + i][key] = vv[i];
    }
    WAIT_ASYNC0();
#else
    for (int t = threadIdx.x; t < 512; t += 256) {
      const int key = t >> 4;
      const int c8  = (t & 15) * 8;
      *(v8h*)&Ks[key][c8] = *(const v8h*)&K[(size_t)(kb + key) * DIM + h * HD + c8];
      const v8h vv = *(const v8h*)&V[(size_t)(kb + key) * DIM + h * HD + c8];
#pragma unroll
      for (int i = 0; i < 8; ++i) Vt[c8 + i][key] = vv[i];
    }
#endif
    __syncthreads();

    // S(16x32) = Q(16x128) x K^T, two 16-key n-tiles, 4 k-steps of 32.
    v8f s0 = {}, s1 = {};
#pragma unroll
    for (int kf = 0; kf < 4; ++kf) {
      v16h b0 = ld8x2(&Ks[l15][kf * 32 + hl * 16], &Ks[l15][kf * 32 + hl * 16 + 8]);
      v16h b1 = ld8x2(&Ks[16 + l15][kf * 32 + hl * 16], &Ks[16 + l15][kf * 32 + hl * 16 + 8]);
      s0 = __builtin_amdgcn_wmma_f32_16x16x32_f16(false, aq[kf], false, b0, (short)0, s0, false, false);
      s1 = __builtin_amdgcn_wmma_f32_16x16x32_f16(false, aq[kf], false, b1, (short)0, s1, false, false);
    }

    // Online softmax per row (rows striped: vgpr vi, lane-half -> M = hl*8+vi).
#pragma unroll
    for (int vi = 0; vi < 8; ++vi) {
      const int qrow = qw + hl * 8 + vi;
      const float v0 = s0[vi] * sc + ((kb + l15) > qrow ? -1e9f : 0.f);
      const float v1 = s1[vi] * sc + ((kb + 16 + l15) > qrow ? -1e9f : 0.f);
      float mx = fmaxf(v0, v1);
#pragma unroll
      for (int off = 1; off < 16; off <<= 1) mx = fmaxf(mx, __shfl_xor(mx, off, 32));
      const float mnew = fmaxf(mrow[vi], mx);
      const float corr = __expf(mrow[vi] - mnew);
      const float p0 = __expf(v0 - mnew);
      const float p1 = __expf(v1 - mnew);
      float rs = p0 + p1;
#pragma unroll
      for (int off = 1; off < 16; off <<= 1) rs += __shfl_xor(rs, off, 32);
      lrow[vi] = lrow[vi] * corr + rs;
      mrow[vi] = mnew;
#pragma unroll
      for (int dt = 0; dt < 8; ++dt) ov[dt][vi] *= corr;
      Ps[wv][hl * 8 + vi][l15]      = (_Float16)p0;
      Ps[wv][hl * 8 + vi][16 + l15] = (_Float16)p1;
    }
    // In-wave LDS RAW: lanes read other lanes' P writes; wave32 lockstep,
    // only need the DS counter drained (no barrier).
    asm volatile("s_wait_dscnt 0" ::: "memory");

    // O(16x128) += P(16x32) x V(32x128): 8 d-tiles.
    v16h pa = ld8x2(&Ps[wv][l15][hl * 16], &Ps[wv][l15][hl * 16 + 8]);
#pragma unroll
    for (int dt = 0; dt < 8; ++dt) {
      v16h bv = ld8x2(&Vt[dt * 16 + l15][hl * 16], &Vt[dt * 16 + l15][hl * 16 + 8]);
      ov[dt] = __builtin_amdgcn_wmma_f32_16x16x32_f16(false, pa, false, bv, (short)0, ov[dt], false, false);
    }
  }

#pragma unroll
  for (int vi = 0; vi < 8; ++vi) {
    const float inv = 1.0f / lrow[vi];
    const size_t row = (size_t)(qw + hl * 8 + vi);
#pragma unroll
    for (int dt = 0; dt < 8; ++dt)
      O[row * DIM + h * HD + dt * 16 + l15] = (_Float16)(ov[dt][vi] * inv);
  }
}

extern "C" void kernel_launch(void* const* d_in, const int* in_sizes, int n_in,
                              void* d_out, int out_size, void* d_ws, size_t ws_size,
                              hipStream_t stream) {
  (void)in_sizes; (void)n_in; (void)out_size; (void)ws_size;
  const float*   x    = (const float*)d_in[0];
  const int8_t*  wqw  = (const int8_t*)d_in[1];
  const float*   wqs  = (const float*)d_in[2];
  const int8_t*  wkw  = (const int8_t*)d_in[3];
  const float*   wks  = (const float*)d_in[4];
  const int8_t*  wvw  = (const int8_t*)d_in[5];
  const float*   wvs  = (const float*)d_in[6];
  const int8_t*  wow  = (const int8_t*)d_in[7];
  const float*   wos  = (const float*)d_in[8];
  const float*   cosp = (const float*)d_in[9];
  const float*   sinp = (const float*)d_in[10];
  float* out = (float*)d_out;

  char* ws = (char*)d_ws;
  _Float16* xh = (_Float16*)(ws);                         // 8 MB
  _Float16* qh = (_Float16*)(ws + ((size_t)8  << 20));    // 8 MB
  _Float16* kh = (_Float16*)(ws + ((size_t)16 << 20));    // 8 MB
  _Float16* vh = (_Float16*)(ws + ((size_t)24 << 20));    // 8 MB
  _Float16* oh = (_Float16*)(ws + ((size_t)32 << 20));    // 8 MB

  const int n = SEQ * DIM;
  cvt_kernel<<<n / 256, 256, 0, stream>>>(x, xh, n);

  dim3 gg(DIM / 128, SEQ / 128);
  gemm_q40_kernel<<<gg, 256, 0, stream>>>(xh, wqw, wqs, qh, nullptr);
  gemm_q40_kernel<<<gg, 256, 0, stream>>>(xh, wkw, wks, kh, nullptr);
  gemm_q40_kernel<<<gg, 256, 0, stream>>>(xh, wvw, wvs, vh, nullptr);

  rope_kernel<<<SEQ * NH * 64 / 256, 256, 0, stream>>>(qh, kh, cosp, sinp);

  attn_kernel<<<dim3(SEQ / 128, NH), 256, 0, stream>>>(qh, kh, vh, oh);

  gemm_q40_kernel<<<gg, 256, 0, stream>>>(oh, wow, wos, nullptr, out);
}
